// SelfAttention_1864015806622
// MI455X (gfx1250) — matrix operand order repaired and checked by reading, not compiled, not run on hardware
//
#include <hip/hip_runtime.h>
#include <hip/hip_bf16.h>

#define SEQ 2048
#define BATCH 2
#define HIDDEN 1024
#define HEADS 16
#define HEAD_DIM 64
#define NTOK (SEQ * BATCH)   /* 4096 */

typedef __bf16 bf16;
typedef __attribute__((ext_vector_type(16))) __bf16 v16bf;
typedef __attribute__((ext_vector_type(8)))  float  v8f;

__device__ inline v8f vzero8() {
    v8f z;
#pragma unroll
    for (int i = 0; i < 8; ++i) z[i] = 0.0f;
    return z;
}

__device__ inline v8f wmma_bf16(v16bf a, v16bf b, v8f c) {
    // v_wmma_f32_16x16x32_bf16: D = A(16x32) * B(32x16) + C(16x16 f32)
    return __builtin_amdgcn_wmma_f32_16x16x32_bf16(
        /*neg_a=*/false, a, /*neg_b=*/false, b,
        /*c_mod=*/(short)0, c, /*reuse_a=*/false, /*reuse_b=*/false);
}

// ---- VALU-only 16-lane row reductions via DPP ROW_XMASK (0x160|mask) -------
template <int MASK>
__device__ inline float row_xor_f32(float v) {
    const int r = __builtin_amdgcn_update_dpp(
        0, __builtin_bit_cast(int, v), 0x160 | MASK, 0xf, 0xf, true);
    return __builtin_bit_cast(float, r);
}
__device__ inline float rowmax16(float v) {
    v = fmaxf(v, row_xor_f32<1>(v));
    v = fmaxf(v, row_xor_f32<2>(v));
    v = fmaxf(v, row_xor_f32<4>(v));
    v = fmaxf(v, row_xor_f32<8>(v));
    return v;
}
__device__ inline float rowsum16(float v) {
    v += row_xor_f32<1>(v);
    v += row_xor_f32<2>(v);
    v += row_xor_f32<4>(v);
    v += row_xor_f32<8>(v);
    return v;
}

// ---- async global -> LDS copy (CDNA5 path, ASYNCcnt-tracked) ---------------
__device__ inline void async_load_b128(unsigned lds_byte_addr,
                                       const bf16* gbase, int byte_off) {
    asm volatile("global_load_async_to_lds_b128 %0, %1, %2"
                 :: "v"(lds_byte_addr), "v"(byte_off), "s"(gbase)
                 : "memory");
}
__device__ inline void wait_async(int n) {
    if (n == 0) asm volatile("s_wait_asynccnt 0" ::: "memory");
    else        asm volatile("s_wait_asynccnt 2" ::: "memory");
}

// A fragment: 16x32 bf16 tile, row-major with leading dim ld.
// ISA layout: row M = lane%16; K(e) = e%8 + 16*(e/8) + 8*(lane/16)
__device__ inline v16bf load_frag_a(const bf16* base, int ld, int lane) {
    const int row = lane & 15, half = lane >> 4;
    v16bf a;
#pragma unroll
    for (int e = 0; e < 16; ++e) {
        const int k = (e & 7) + ((e >> 3) << 4) + (half << 3);
        a[e] = base[row * ld + k];
    }
    return a;
}

// B fragment (K=32) from a tile stored [N][K] row-major (computes A*T^T).
// ISA layout: col N = lane%16; K(e) = e%8 + 16*(e/8) + 8*(lane/16)
__device__ inline v16bf load_frag_bt(const bf16* base, int ld, int lane) {
    const int n = lane & 15, half = lane >> 4;
    v16bf b;
#pragma unroll
    for (int e = 0; e < 16; ++e) {
        const int k = (e & 7) + ((e >> 3) << 4) + (half << 3);
        b[e] = base[n * ld + k];
    }
    return b;
}

// B fragment (K=32) from a tile stored [K][N] row-major.
__device__ inline v16bf load_frag_b(const bf16* base, int ld, int lane) {
    const int n = lane & 15, half = lane >> 4;
    v16bf b;
#pragma unroll
    for (int e = 0; e < 16; ++e) {
        const int k = (e & 7) + ((e >> 3) << 4) + (half << 3);
        b[e] = base[k * ld + n];
    }
    return b;
}

// ---------------------------------------------------------------------------
// GEMM: Out[M,N] = ( A[M,K] @ W[N,K]^T + bias ) * scale.
// A_BF16: A is bf16 (ctx) else fp32 (hidden_states, converted while staging).
// OUT_QKV: scatter bf16 into [B,HEADS,SEQ,HEAD_DIM]; else fp32 token-major.
// ---------------------------------------------------------------------------
template <bool A_BF16, bool OUT_QKV>
__global__ void __launch_bounds__(256)
gemm_xwt_kernel(const void* __restrict__ Av, const float* __restrict__ W,
                const float* __restrict__ bias, void* __restrict__ Outv,
                int N, int K, float scale) {
    __shared__ __align__(16) bf16 sA[64][40];
    __shared__ __align__(16) bf16 sW[64][40];
    const int tid = threadIdx.x;
    const int wave = tid >> 5, lane = tid & 31;
    const int m0 = blockIdx.x * 64, n0 = blockIdx.y * 64;
    // 8 waves cover the 64x64 tile: row strip (wave/2)*16, col strip (wave&1)*32
    const int tr  = (wave >> 1) * 16;
    const int tc0 = (wave & 1) * 32;
    const int tc1 = tc0 + 16;
    v8f acc0 = vzero8(), acc1 = vzero8();
    const int lr = tid >> 2;        // 0..63
    const int lc = (tid & 3) * 8;   // 0,8,16,24

    for (int k0 = 0; k0 < K; k0 += 32) {
        __syncthreads();
        if constexpr (A_BF16) {
            const bf16* A = (const bf16*)Av;
#pragma unroll
            for (int j = 0; j < 8; ++j)
                sA[lr][lc + j] = A[(size_t)(m0 + lr) * K + k0 + lc + j];
        } else {
            const float* A = (const float*)Av;
#pragma unroll
            for (int j = 0; j < 8; ++j)
                sA[lr][lc + j] = (bf16)A[(size_t)(m0 + lr) * K + k0 + lc + j];
        }
#pragma unroll
        for (int j = 0; j < 8; ++j)
            sW[lr][lc + j] = (bf16)W[(size_t)(n0 + lr) * K + k0 + lc + j];
        __syncthreads();

        const v16bf a  = load_frag_a(&sA[tr][0], 40, lane);
        const v16bf b0 = load_frag_bt(&sW[tc0][0], 40, lane);
        const v16bf b1 = load_frag_bt(&sW[tc1][0], 40, lane);
        acc0 = wmma_bf16(a, b0, acc0);
        acc1 = wmma_bf16(a, b1, acc1);
    }

    const int col = lane & 15, half = lane >> 4;
#pragma unroll
    for (int r = 0; r < 8; ++r) {
        const int gm  = m0 + tr + r + half * 8;       // token index t = s*BATCH + b
        const int gn0 = n0 + tc0 + col;
        const int gn1 = n0 + tc1 + col;
        const float v0 = (acc0[r] + bias[gn0]) * scale;
        const float v1 = (acc1[r] + bias[gn1]) * scale;
        if constexpr (OUT_QKV) {
            bf16* Out = (bf16*)Outv;
            const int s = gm / BATCH, b = gm % BATCH;
            {
                const int h = gn0 / HEAD_DIM, d = gn0 % HEAD_DIM;
                Out[(((size_t)(b * HEADS + h) * SEQ) + s) * HEAD_DIM + d] = (bf16)v0;
            }
            {
                const int h = gn1 / HEAD_DIM, d = gn1 % HEAD_DIM;
                Out[(((size_t)(b * HEADS + h) * SEQ) + s) * HEAD_DIM + d] = (bf16)v1;
            }
        } else {
            float* Out = (float*)Outv;
            Out[(size_t)gm * N + gn0] = v0;
            Out[(size_t)gm * N + gn1] = v1;
        }
    }
}

// ---------------------------------------------------------------------------
// Causal flash attention. Grid: (SEQ/128, BATCH*HEADS), 256 threads = 8 waves.
// Wave w handles query rows [blk*128 + 16w, +16). Q pre-scaled by 1/sqrt(d).
// K/V tiles double-buffered via GLOBAL_LOAD_ASYNC_TO_LDS_B128: block i+1's
// DMA overlaps block i's WMMAs; s_wait_asynccnt 2 gates only the live block.
// ---------------------------------------------------------------------------
__global__ void __launch_bounds__(256)
attn_kernel(const bf16* __restrict__ Q, const bf16* __restrict__ Kc,
            const bf16* __restrict__ Vc, bf16* __restrict__ ctx) {
    __shared__ __align__(16) bf16 sK0[32][72];   // row stride 144 B (16B-aligned)
    __shared__ __align__(16) bf16 sV0[32][72];
    __shared__ __align__(16) bf16 sK1[32][72];
    __shared__ __align__(16) bf16 sV1[32][72];
    __shared__ __align__(16) bf16 sP[8][16 * 32];

    const int tid = threadIdx.x;
    const int wave = tid >> 5, lane = tid & 31;
    const int bh = blockIdx.y;                       // b*HEADS + h
    const bf16* Qh = Q  + (size_t)bh * SEQ * HEAD_DIM;
    const bf16* Kh = Kc + (size_t)bh * SEQ * HEAD_DIM;
    const bf16* Vh = Vc + (size_t)bh * SEQ * HEAD_DIM;
    const int qtile = blockIdx.x * 128;
    const int qbase = qtile + wave * 16;
    const int row = lane & 15, half = lane >> 4;
    const int col = row;

    // Q fragments: d 0..31 and d 32..63
    v16bf aq0, aq1;
#pragma unroll
    for (int e = 0; e < 16; ++e) {
        const int d = (e & 7) + ((e >> 3) << 4) + (half << 3);
        aq0[e] = Qh[(size_t)(qbase + row) * HEAD_DIM + d];
        aq1[e] = Qh[(size_t)(qbase + row) * HEAD_DIM + 32 + d];
    }

    float m[8], l[8];
    v8f acc0 = vzero8(), acc1 = vzero8(), acc2 = vzero8(), acc3 = vzero8();
#pragma unroll
    for (int r = 0; r < 8; ++r) { m[r] = -1e30f; l[r] = 0.0f; }

    // async-copy addressing: thread -> (row = tid/8, 16-byte chunk = tid%8)
    const int lrk = tid >> 3;                 // 0..31
    const int lbo = (tid & 7) * 16;           // byte offset within 128B row
    const unsigned doff = (unsigned)(lrk * 144 + lbo);
    const unsigned kdst0 = (unsigned)(size_t)(&sK0[0][0]) + doff;
    const unsigned vdst0 = (unsigned)(size_t)(&sV0[0][0]) + doff;
    const unsigned kdst1 = (unsigned)(size_t)(&sK1[0][0]) + doff;
    const unsigned vdst1 = (unsigned)(size_t)(&sV1[0][0]) + doff;

    const int kend = qtile + 128;             // uniform across the WG

    // prologue: start DMA of block 0 into buffer 0
    {
        const int go = lrk * (HEAD_DIM * 2) + lbo;
        async_load_b128(kdst0, Kh, go);
        async_load_b128(vdst0, Vh, go);
    }

    for (int kb = 0, it = 0; kb < kend; kb += 32, ++it) {
        const int bi = it & 1;
        const bf16* cK = bi ? &sK1[0][0] : &sK0[0][0];
        const bf16* cV = bi ? &sV1[0][0] : &sV0[0][0];
        // 1) everyone finished reading the other buffer -> safe to DMA into it
        __syncthreads();
        const bool more = (kb + 32 < kend);
        if (more) {
            const int go = (kb + 32 + lrk) * (HEAD_DIM * 2) + lbo;
            async_load_b128(bi ? kdst0 : kdst1, Kh, go);
            async_load_b128(bi ? vdst0 : vdst1, Vh, go);
            if (kb + 64 < kend) {   // warm GL2 two blocks ahead
                __builtin_prefetch(Kh + (size_t)(kb + 64) * HEAD_DIM + tid * 8, 0, 1);
                __builtin_prefetch(Vh + (size_t)(kb + 64) * HEAD_DIM + tid * 8, 0, 1);
            }
            wait_async(2);          // current block landed; next block in flight
        } else {
            wait_async(0);
        }
        // 2) current block visible to all waves
        __syncthreads();

        // scores S = Q * K^T, key groups g=0 (keys kb..kb+15) and g=1
        v8f s0 = vzero8(), s1 = vzero8();
        s0 = wmma_bf16(aq0, load_frag_bt(cK + 0 * 72 + 0,   72, lane), s0);
        s0 = wmma_bf16(aq1, load_frag_bt(cK + 0 * 72 + 32,  72, lane), s0);
        s1 = wmma_bf16(aq0, load_frag_bt(cK + 16 * 72 + 0,  72, lane), s1);
        s1 = wmma_bf16(aq1, load_frag_bt(cK + 16 * 72 + 32, 72, lane), s1);

        // causal mask + online softmax (rows live within a 16-lane half)
#pragma unroll
        for (int r = 0; r < 8; ++r) {
            const int qrow = qbase + r + half * 8;
            const float v0 = (kb + col      <= qrow) ? s0[r] : -1e30f;
            const float v1 = (kb + 16 + col <= qrow) ? s1[r] : -1e30f;
            const float mx = rowmax16(fmaxf(v0, v1));
            const float mn = fmaxf(m[r], mx);
            const float corr = __expf(m[r] - mn);
            m[r] = mn;
            const float p0 = __expf(v0 - mn);
            const float p1 = __expf(v1 - mn);
            const float rs = rowsum16(p0 + p1);
            l[r] = l[r] * corr + rs;
            acc0[r] *= corr; acc1[r] *= corr; acc2[r] *= corr; acc3[r] *= corr;
            bf16* prow = &sP[wave][(r + half * 8) * 32];
            prow[col]      = (bf16)p0;
            prow[16 + col] = (bf16)p1;
        }
        __syncthreads();

        // ctx += P(16x32) * V(32x64), d-chunks of 16
        const v16bf ap = load_frag_a(&sP[wave][0], 32, lane);
        acc0 = wmma_bf16(ap, load_frag_b(cV + 0,  72, lane), acc0);
        acc1 = wmma_bf16(ap, load_frag_b(cV + 16, 72, lane), acc1);
        acc2 = wmma_bf16(ap, load_frag_b(cV + 32, 72, lane), acc2);
        acc3 = wmma_bf16(ap, load_frag_b(cV + 48, 72, lane), acc3);
    }

    // epilogue: ctx token-major bf16 [s*BATCH+b][HIDDEN]
    const int b = bh / HEADS, h = bh % HEADS;
#pragma unroll
    for (int r = 0; r < 8; ++r) {
        const int s = qbase + r + half * 8;
        const size_t tbase = ((size_t)s * BATCH + b) * HIDDEN + (size_t)h * HEAD_DIM;
        const float inv = 1.0f / l[r];
        ctx[tbase + 0  + col] = (bf16)(acc0[r] * inv);
        ctx[tbase + 16 + col] = (bf16)(acc1[r] * inv);
        ctx[tbase + 32 + col] = (bf16)(acc2[r] * inv);
        ctx[tbase + 48 + col] = (bf16)(acc3[r] * inv);
    }
}

extern "C" void kernel_launch(void* const* d_in, const int* in_sizes, int n_in,
                              void* d_out, int out_size, void* d_ws, size_t ws_size,
                              hipStream_t stream) {
    const float* hs = (const float*)d_in[0];
    // d_in[1] = attention_mask (causal tril) — implemented arithmetically
    const float* Wq = (const float*)d_in[2];
    const float* bq = (const float*)d_in[3];
    const float* Wk = (const float*)d_in[4];
    const float* bk = (const float*)d_in[5];
    const float* Wv = (const float*)d_in[6];
    const float* bv = (const float*)d_in[7];
    const float* Wo = (const float*)d_in[8];
    const float* bo = (const float*)d_in[9];
    float* out = (float*)d_out;

    bf16* q   = (bf16*)d_ws;                           // [B,H,S,D] bf16
    bf16* kq  = q  + (size_t)NTOK * HIDDEN;
    bf16* vq  = kq + (size_t)NTOK * HIDDEN;
    bf16* ctx = vq + (size_t)NTOK * HIDDEN;            // [NTOK,HIDDEN] bf16

    const dim3 blk(256);
    const dim3 ggrid(NTOK / 64, HIDDEN / 64);
    const float inv_norm = 0.125f;                     // 1/sqrt(64)

    gemm_xwt_kernel<false, true><<<ggrid, blk, 0, stream>>>(hs, Wq, bq, q,  HIDDEN, HIDDEN, inv_norm);
    gemm_xwt_kernel<false, true><<<ggrid, blk, 0, stream>>>(hs, Wk, bk, kq, HIDDEN, HIDDEN, 1.0f);
    gemm_xwt_kernel<false, true><<<ggrid, blk, 0, stream>>>(hs, Wv, bv, vq, HIDDEN, HIDDEN, 1.0f);

    const dim3 agrid(SEQ / 128, BATCH * HEADS);
    attn_kernel<<<agrid, blk, 0, stream>>>(q, kq, vq, ctx);

    gemm_xwt_kernel<true, false><<<ggrid, blk, 0, stream>>>(ctx, Wo, bo, out, HIDDEN, HIDDEN, 1.0f);
}
